// MetaTimeTrajTypeGRUNet_12652973654360
// MI455X (gfx1250) — matrix operand order repaired and checked by reading, not compiled, not run on hardware
//
#include <hip/hip_runtime.h>
#include <hip/hip_bf16.h>

// ---------------------------------------------------------------------------
// MetaTimeTrajTypeGRUNet forward for MI455X (gfx1250, wave32, WMMA f16->f32)
//
//  Pipeline:
//    prep:   f32 -> f16 weight repack (w_ih||w_hh combined, fc_w padded)
//    gru<L>: 2 persistent workgroups (16 batch rows each), T-step loop,
//            per step M=16 x N=1536 x K=(KXP+512) GEMM with v_wmma 16x16x32;
//            layer-1 x-panel staged via global_load_async_to_lds_b128
//    fc:     [8192x512] @ [512x10016] wmma GEMM, 2Mx2N register tiling,
//            async-LDS A staging; raw logits -> d_out, sigmoid(time) direct
//    stats:  per-row max + log-sum-exp (float4)
//    sub:    log_softmax in place (float4)
// ---------------------------------------------------------------------------

typedef __attribute__((ext_vector_type(16))) _Float16 v16h;
typedef __attribute__((ext_vector_type(8)))  _Float16 v8h;
typedef __attribute__((ext_vector_type(8)))  float    v8f;

#define SEQ   256
#define BATCH 32
#define EMB   256
#define HID   512
#define NOUT  10001
#define NOUTP 10016         // padded to 16
#define NTILES (NOUTP / 16) // 626
#define LOC_ELEMS (32u * 256u * 10000u)   // 81,920,000

__device__ __forceinline__ float sigmoidf_(float x) { return 1.f / (1.f + __expf(-x)); }

// A fragment 16x32 f16 from LDS (row-major, 'stride' halfs per row).
// Contiguous k order, same as ldB: lanes 0-15 -> row M=lane, K 0..15; lanes 16-31 -> K 16..31
__device__ __forceinline__ v16h ldA(const _Float16* panel, int stride, int lane, int k0) {
    int row = lane & 15;
    int kb  = (lane & 16) ? 16 : 0;
    const _Float16* p = panel + row * stride + k0 + kb;
    v8h lo = *(const v8h*)p;          // ds_load_b128
    v8h hi = *(const v8h*)(p + 8);    // ds_load_b128
    v16h a;
#pragma unroll
    for (int i = 0; i < 8; ++i) { a[i] = lo[i]; a[i + 8] = hi[i]; }
    return a;
}

// B fragment 32x16 f16 from global weights stored [N][K] row-major.
// Lanes 0-15 -> col N=lane, K 0..15; lanes 16-31 -> col N=lane-16, K 16..31
__device__ __forceinline__ v16h ldB(const _Float16* W, int ldw, int n0, int k0, int lane) {
    const _Float16* p = W + (size_t)(n0 + (lane & 15)) * ldw + k0 + ((lane & 16) ? 16 : 0);
    return *(const v16h*)p;           // global_load_b128 x2
}

__device__ __forceinline__ v8f wmma16(v16h a, v16h b, v8f c) {
    return __builtin_amdgcn_wmma_f32_16x16x32_f16(false, a, false, b, (short)0, c, false, false);
}

// async copy of one 16-byte chunk: global -> LDS (ASYNCcnt-tracked)
__device__ __forceinline__ void async_g2l_b128(uint32_t lds_addr, uint64_t gaddr) {
    asm volatile("global_load_async_to_lds_b128 %0, %1, off"
                 :: "v"(lds_addr), "v"(gaddr) : "memory");
}
__device__ __forceinline__ void wait_async0() {
    asm volatile("s_wait_asynccnt 0x0" ::: "memory");
}

// ------------------------------ prep kernels ------------------------------

__global__ void prep_wc(const float* __restrict__ wih, const float* __restrict__ whh,
                        int KX, int KXP, _Float16* __restrict__ dst) {
    const int KT = KXP + HID;
    const int total = 3 * HID * KT;
    for (int idx = blockIdx.x * blockDim.x + threadIdx.x; idx < total;
         idx += gridDim.x * blockDim.x) {
        int n = idx / KT, k = idx - n * KT;
        float v;
        if (k < KX)       v = wih[n * KX + k];
        else if (k < KXP) v = 0.f;
        else              v = whh[n * HID + (k - KXP)];
        dst[idx] = (_Float16)v;
    }
}

__global__ void prep_fcw(const float* __restrict__ fcw, _Float16* __restrict__ dst) {
    const int total = NOUTP * HID;
    for (int idx = blockIdx.x * blockDim.x + threadIdx.x; idx < total;
         idx += gridDim.x * blockDim.x) {
        int n = idx >> 9, k = idx & 511;
        dst[idx] = (_Float16)((n < NOUT) ? fcw[n * HID + k] : 0.f);
    }
}

// ------------------------------ GRU kernel --------------------------------
// Persistent: grid.x = 2 (batch halves), block = 256 threads (8 waves).
// Wave w owns hidden columns [64w, 64w+64): 4 tiles x {r,z,n} gates.

template <int LAYER, int KX, int KXP>
__global__ __launch_bounds__(256)
void gru_kernel(const int*      __restrict__ locations,
                const float*    __restrict__ times,
                const float*    __restrict__ emb,
                const float*    __restrict__ traj_emb,
                const int*      __restrict__ labels,
                const _Float16* __restrict__ xin,   // layer1: y0 [T][B][HID] f16
                const _Float16* __restrict__ Wc,    // [1536][KXP+HID] f16
                const float*    __restrict__ b_ih,
                const float*    __restrict__ b_hh,
                _Float16*       __restrict__ yout) {
    constexpr int KTOT = KXP + HID;
    __shared__ _Float16 Apanel[16][KTOT];       // [x_t || h] f16
    __shared__ float    hbuf[2][16][HID];       // f32 hidden state, ping-pong

    const int tid = threadIdx.x;
    const int b0  = blockIdx.x * 16;
    const int lane = tid & 31, w = tid >> 5;

    // init hidden: reproduce torch .view scramble of traj_emb[labels]
    for (int idx = tid; idx < 16 * HID; idx += 256) {
        int m = idx >> 9, k = idx & 511;
        int mg = b0 + m;
        int sb = LAYER * 16 + (mg >> 1);
        int sj = ((mg & 1) << 9) + k;
        float h0 = traj_emb[(size_t)labels[sb] * (2 * HID) + sj];
        hbuf[0][m][k] = h0;
        Apanel[m][KXP + k] = (_Float16)h0;
    }
    if constexpr (KXP > KX) {   // zero pad columns [KX, KXP) once
        for (int idx = tid; idx < 16 * (KXP - KX); idx += 256) {
            int m = idx / (KXP - KX), k = KX + idx % (KXP - KX);
            Apanel[m][k] = (_Float16)0.f;
        }
    }
    __syncthreads();

    for (int t = 0; t < SEQ; ++t) {
        const int cur = t & 1, nxt = cur ^ 1;
        // -------- build x-section of A panel --------
        if constexpr (LAYER == 0) {
            for (int idx = tid; idx < 16 * EMB; idx += 256) {
                int m = idx >> 8, k = idx & 255;
                int loc = locations[(b0 + m) * SEQ + t];
                Apanel[m][k] = (_Float16)emb[(size_t)loc * EMB + k];
            }
            if (tid < 16) Apanel[tid][EMB] = (_Float16)times[(b0 + tid) * SEQ + t];
        } else {
            // async DMA: 16 rows x 1024B from y0 into strided x-section
            uint32_t lds0 = (uint32_t)(uintptr_t)&Apanel[0][0];
            uint64_t g0   = (uint64_t)(uintptr_t)(xin + (size_t)(t * BATCH + b0) * HID);
#pragma unroll
            for (int i = 0; i < 4; ++i) {
                int c = tid + i * 256;            // 1024 chunks of 16B
                int m = c >> 6;                   // 64 chunks per row
                int ko = (c & 63) * 16;
                async_g2l_b128(lds0 + (uint32_t)m * (KTOT * 2) + (uint32_t)ko,
                               g0 + (uint64_t)c * 16);
            }
            wait_async0();
        }
        __syncthreads();

        // -------- gate GEMMs + elementwise update, per 16-col tile --------
#pragma unroll
        for (int j = 0; j < 4; ++j) {
            const int col = w * 64 + j * 16;
            v8f sr = {}, sz = {}, xn = {}, hn = {};
            for (int k0 = 0; k0 < KTOT; k0 += 32) {
                v16h a  = ldA(&Apanel[0][0], KTOT, lane, k0);
                v16h br = ldB(Wc, KTOT, col,            k0, lane);
                sr = wmma16(a, br, sr);
                v16h bz = ldB(Wc, KTOT, HID + col,      k0, lane);
                sz = wmma16(a, bz, sz);
                v16h bn = ldB(Wc, KTOT, 2 * HID + col,  k0, lane);
                if (k0 < KXP) xn = wmma16(a, bn, xn);   // input-proj part
                else          hn = wmma16(a, bn, hn);   // hidden-proj part
            }
            const int   cl  = col + (lane & 15);
            const float brr = b_ih[cl]            + b_hh[cl];
            const float bzz = b_ih[HID + cl]      + b_hh[HID + cl];
            const float bxn = b_ih[2 * HID + cl];
            const float bhn = b_hh[2 * HID + cl];
#pragma unroll
            for (int e = 0; e < 8; ++e) {
                const int row = e + ((lane >> 4) << 3);
                float r  = sigmoidf_(sr[e] + brr);
                float z  = sigmoidf_(sz[e] + bzz);
                float n  = tanhf(xn[e] + bxn + r * (hn[e] + bhn));
                float hp = hbuf[cur][row][cl];
                float hv = (1.f - z) * n + z * hp;
                hbuf[nxt][row][cl] = hv;
                float yv = (LAYER == 1) ? fmaxf(hv, 0.f) : hv;   // relu before FC
                size_t oidx = (LAYER == 0)
                    ? ((size_t)(t * BATCH + b0 + row) * HID + cl)   // [T][B][H]
                    : ((size_t)((b0 + row) * SEQ + t) * HID + cl);  // [B][T][H]
                yout[oidx] = (_Float16)yv;
            }
        }
        __syncthreads();
        // refresh h-section of A panel for next step
        for (int idx = tid; idx < 16 * HID; idx += 256) {
            int m = idx >> 9, k = idx & 511;
            Apanel[m][KXP + k] = (_Float16)hbuf[nxt][m][k];
        }
        __syncthreads();
    }
}

// ------------------------------ FC kernel ---------------------------------
// grid: (ceil(626/16)=40, 8192/32=256); block 256 (8 waves).
// Block: 32 rows (2 M-tiles) staged in LDS via async DMA; wave -> 2 N-tiles.
// 2Mx2N register tile: 4 wmma per (2 A + 2 B) fragment loads.

__global__ __launch_bounds__(256)
void fc_kernel(const _Float16* __restrict__ y1h,
               const _Float16* __restrict__ fcw,
               const float*    __restrict__ fcb,
               float*          __restrict__ out_loc,
               float*          __restrict__ out_time) {
    __shared__ _Float16 Ap[32][HID];    // 32 KB
    const int tid = threadIdx.x;
    const int mb  = blockIdx.y;
    {   // async-stage contiguous 32x512 f16 = 2048 x 16B chunks
        uint32_t lds0 = (uint32_t)(uintptr_t)&Ap[0][0];
        uint64_t g0   = (uint64_t)(uintptr_t)(y1h + (size_t)mb * 32 * HID);
#pragma unroll
        for (int i = 0; i < 8; ++i) {
            int c = tid + i * 256;
            async_g2l_b128(lds0 + (uint32_t)c * 16, g0 + (uint64_t)c * 16);
        }
        wait_async0();
    }
    __syncthreads();

    const int lane = tid & 31, w = tid >> 5;
    const int nt0 = blockIdx.x * 16 + w * 2;
    const int nt1 = nt0 + 1;
    if (nt0 >= NTILES) return;          // wave-uniform
    const bool has1 = (nt1 < NTILES);   // wave-uniform

    v8f a00 = {}, a01 = {}, a10 = {}, a11 = {};
    for (int k0 = 0; k0 < HID; k0 += 32) {
        v16h a0 = ldA(&Ap[0][0],  HID, lane, k0);
        v16h a1 = ldA(&Ap[16][0], HID, lane, k0);
        v16h b0 = ldB(fcw, HID, nt0 * 16, k0, lane);
        a00 = wmma16(a0, b0, a00);
        a10 = wmma16(a1, b0, a10);
        if (has1) {
            v16h b1 = ldB(fcw, HID, nt1 * 16, k0, lane);
            a01 = wmma16(a0, b1, a01);
            a11 = wmma16(a1, b1, a11);
        }
    }

    auto emit = [&](int nt, int mi, const v8f& acc) {
        const int   n    = nt * 16 + (lane & 15);
        const float bias = (n < NOUT) ? fcb[n] : 0.f;
#pragma unroll
        for (int e = 0; e < 8; ++e) {
            const int row = mb * 32 + mi * 16 + e + ((lane >> 4) << 3);  // b*T + t
            float v = acc[e] + bias;
            if (n < NOUT - 1)       out_loc[(size_t)row * 10000 + n] = v;
            else if (n == NOUT - 1) out_time[row] = sigmoidf_(v);
        }
    };
    emit(nt0, 0, a00);
    emit(nt0, 1, a10);
    if (has1) { emit(nt1, 0, a01); emit(nt1, 1, a11); }
}

// --------------------------- softmax kernels ------------------------------

__global__ __launch_bounds__(256)
void softmax_stats(const float* __restrict__ loc, float* __restrict__ stats) {
    __shared__ float red[256];
    const int r = blockIdx.x, tid = threadIdx.x;
    const float4* row4 = (const float4*)(loc + (size_t)r * 10000);   // 2500 vec4
    float m = -3.4e38f;
    for (int i = tid; i < 2500; i += 256) {
        float4 v = row4[i];
        m = fmaxf(m, fmaxf(fmaxf(v.x, v.y), fmaxf(v.z, v.w)));
    }
    red[tid] = m; __syncthreads();
    for (int s = 128; s > 0; s >>= 1) { if (tid < s) red[tid] = fmaxf(red[tid], red[tid + s]); __syncthreads(); }
    m = red[0]; __syncthreads();
    float sum = 0.f;
    for (int i = tid; i < 2500; i += 256) {
        float4 v = row4[i];
        sum += __expf(v.x - m) + __expf(v.y - m) + __expf(v.z - m) + __expf(v.w - m);
    }
    red[tid] = sum; __syncthreads();
    for (int s = 128; s > 0; s >>= 1) { if (tid < s) red[tid] += red[tid + s]; __syncthreads(); }
    if (tid == 0) stats[r] = m + logf(red[0]);
}

__global__ __launch_bounds__(256)
void softmax_sub(float* __restrict__ loc, const float* __restrict__ stats) {
    const int r = blockIdx.x;
    const float s = stats[r];
    float4* row4 = (float4*)(loc + (size_t)r * 10000);
    for (int i = threadIdx.x; i < 2500; i += 256) {
        float4 v = row4[i];
        v.x -= s; v.y -= s; v.z -= s; v.w -= s;
        row4[i] = v;
    }
}

// ------------------------------ launcher ----------------------------------

extern "C" void kernel_launch(void* const* d_in, const int* in_sizes, int n_in,
                              void* d_out, int out_size, void* d_ws, size_t ws_size,
                              hipStream_t stream) {
    const int*   locations = (const int*)  d_in[0];
    const float* times     = (const float*)d_in[1];
    const int*   labels    = (const int*)  d_in[2];
    const float* emb_table = (const float*)d_in[3];
    const float* traj_emb  = (const float*)d_in[4];
    const float* w_ih0     = (const float*)d_in[5];
    const float* w_hh0     = (const float*)d_in[6];
    const float* b_ih0     = (const float*)d_in[7];
    const float* b_hh0     = (const float*)d_in[8];
    const float* w_ih1     = (const float*)d_in[9];
    const float* w_hh1     = (const float*)d_in[10];
    const float* b_ih1     = (const float*)d_in[11];
    const float* b_hh1     = (const float*)d_in[12];
    const float* fc_w      = (const float*)d_in[13];
    const float* fc_b      = (const float*)d_in[14];

    float* out_loc  = (float*)d_out;
    float* out_time = (float*)d_out + LOC_ELEMS;

    // workspace carve-up (256B aligned), ~33 MB total
    char* p = (char*)d_ws;
    auto carve = [&](size_t bytes) -> char* {
        char* r = p; p += (bytes + 255) & ~(size_t)255; return r;
    };
    _Float16* wc0h  = (_Float16*)carve((size_t)1536 * 800  * 2);  // [1536][288+512]
    _Float16* wc1h  = (_Float16*)carve((size_t)1536 * 1024 * 2);  // [1536][512+512]
    _Float16* fcwh  = (_Float16*)carve((size_t)NOUTP * HID * 2);  // [10016][512]
    _Float16* y0h   = (_Float16*)carve((size_t)SEQ * BATCH * HID * 2);
    _Float16* y1h   = (_Float16*)carve((size_t)SEQ * BATCH * HID * 2);
    float*    stats = (float*)   carve((size_t)BATCH * SEQ * 4);

    // weight repack f32 -> f16
    prep_wc <<<4800, 256, 0, stream>>>(w_ih0, w_hh0, 257, 288, wc0h);
    prep_wc <<<6144, 256, 0, stream>>>(w_ih1, w_hh1, 512, 512, wc1h);
    prep_fcw<<<20032, 256, 0, stream>>>(fc_w, fcwh);

    // GRU layers (2 persistent workgroups each, no cross-WG dependency)
    gru_kernel<0, 257, 288><<<2, 256, 0, stream>>>(
        locations, times, emb_table, traj_emb, labels,
        nullptr, wc0h, b_ih0, b_hh0, y0h);
    gru_kernel<1, 512, 512><<<2, 256, 0, stream>>>(
        nullptr, nullptr, nullptr, traj_emb, labels,
        y0h, wc1h, b_ih1, b_hh1, y1h);

    // FC: logits -> d_out (raw), time sigmoid -> d_out tail
    fc_kernel<<<dim3((NTILES + 15) / 16, (BATCH * SEQ) / 32), 256, 0, stream>>>(
        y1h, fcwh, fc_b, out_loc, out_time);

    // log_softmax in place
    softmax_stats<<<BATCH * SEQ, 256, 0, stream>>>(out_loc, stats);
    softmax_sub  <<<BATCH * SEQ, 256, 0, stream>>>(out_loc, stats);
}
